// TpjBrain_47699906789819
// MI455X (gfx1250) — matrix operand order repaired and checked by reading, not compiled, not run on hardware
//
#include <hip/hip_runtime.h>
#include <math.h>

typedef unsigned short u16;
typedef __attribute__((ext_vector_type(16))) __bf16 v16bf;
typedef __attribute__((ext_vector_type(8)))  float  v8f;

union Frag { v16bf v; unsigned u[8]; };

__device__ inline u16 f32_to_bf16(float x) {
  unsigned u = __float_as_uint(x);
  unsigned r = (u + 0x7FFFu + ((u >> 16) & 1u)) >> 16;
  return (u16)r;
}

__device__ inline float gelu_f(float x) {
  const float c = 0.7978845608028654f;
  float t = tanhf(c * (x + 0.044715f * x * x * x));
  return 0.5f * x * (1.0f + t);
}

__device__ inline v8f wmma_bf16(v16bf a, v16bf b, v8f c) {
  return __builtin_amdgcn_wmma_f32_16x16x32_bf16(false, a, false, b, (short)0, c, false, false);
}

// Async global->LDS copy of 16 bytes per lane (CDNA5 ASYNCcnt path).
__device__ inline void async_g2l_b128(const u16* gaddr, u16* lds_ptr) {
  unsigned loff = (unsigned)(unsigned long long)lds_ptr;  // low 32 bits = LDS offset
  asm volatile("global_load_async_to_lds_b128 %0, %1, off"
               :: "v"(loff), "v"(gaddr) : "memory");
}
#define S_WAIT_ASYNC(n) asm volatile("s_wait_asynccnt " #n ::: "memory")

// A-matrix 16x32 bf16 fragment (M x K), source row-major [m][k], lda in elems.
__device__ inline v16bf frag_a(const u16* base, int lda, int mbase, int kofs, int lane) {
  Frag f;
  int m  = mbase + (lane & 15);
  int kh = (lane >> 4) << 4;
  const u16* row = base + (size_t)m * lda + kofs;
#pragma unroll
  for (int i = 0; i < 8; ++i) {
    int k = kh + (i << 1);
    f.u[i] = *(const unsigned*)(row + k);
  }
  return f.v;
}

// B-matrix 32x16 bf16 fragment (K x N), source stored N-major: base[n][k].
__device__ inline v16bf frag_b_n(const u16* base, int ldb, int nbase, int kofs, int lane) {
  Frag f;
  int n  = nbase + (lane & 15);
  int kb = (lane >> 4) << 4;
  const u16* row = base + (size_t)n * ldb + kofs + kb;
#pragma unroll
  for (int i = 0; i < 8; ++i) f.u[i] = *(const unsigned*)(row + (i << 1));
  return f.v;
}

// B-matrix 32x16 bf16 fragment, source stored K-major: base[k][n].
__device__ inline v16bf frag_b_k(const u16* base, int ldb, int nbase, int lane) {
  Frag f;
  int n  = nbase + (lane & 15);
  int kb = (lane >> 4) << 4;
#pragma unroll
  for (int i = 0; i < 8; ++i) {
    unsigned lo = base[(size_t)(kb + 2 * i) * ldb + n];
    unsigned hi = base[(size_t)(kb + 2 * i + 1) * ldb + n];
    f.u[i] = lo | (hi << 16);
  }
  return f.v;
}

// ---------------- cast + transpose weights: WT[n*K + k] = bf16(W[k*N + n]) ----
__global__ __launch_bounds__(256) void k_cast_tr(const float* __restrict__ W,
                                                 u16* __restrict__ WT,
                                                 int K, int N) {
  int i = blockIdx.x * 256 + threadIdx.x;
  if (i >= K * N) return;
  int k = i / N, n = i - k * N;
  WT[(size_t)n * K + k] = f32_to_bf16(W[i]);
}

// ---------------- embedding: x = wte[idx] + wpe -------------------------------
__global__ __launch_bounds__(256) void k_embed(const int* __restrict__ idx,
                                               const float* __restrict__ wte,
                                               const float* __restrict__ wpe,
                                               float* __restrict__ x,
                                               int T, int E, int total) {
  int i = blockIdx.x * 256 + threadIdx.x;
  if (i >= total) return;
  int e = i % E;
  int bt = i / E;
  int t = bt % T;
  x[i] = wte[(size_t)idx[bt] * E + e] + wpe[(size_t)t * E + e];
}

// ---------------- layernorm -> bf16 ------------------------------------------
__global__ __launch_bounds__(256) void k_layernorm(const float* __restrict__ X,
                                                   const float* __restrict__ w,
                                                   const float* __restrict__ b,
                                                   u16* __restrict__ Out, int E) {
  __shared__ float red[256];
  int row = blockIdx.x;
  int tid = threadIdx.x;
  const float* x = X + (size_t)row * E;
  float s = 0.f;
  for (int e = tid; e < E; e += 256) s += x[e];
  red[tid] = s; __syncthreads();
  for (int off = 128; off > 0; off >>= 1) {
    if (tid < off) red[tid] += red[tid + off];
    __syncthreads();
  }
  float mean = red[0] / (float)E;
  __syncthreads();
  float v = 0.f;
  for (int e = tid; e < E; e += 256) { float d = x[e] - mean; v += d * d; }
  red[tid] = v; __syncthreads();
  for (int off = 128; off > 0; off >>= 1) {
    if (tid < off) red[tid] += red[tid + off];
    __syncthreads();
  }
  float rstd = rsqrtf(red[0] / (float)E + 1e-5f);
  for (int e = tid; e < E; e += 256)
    Out[(size_t)row * E + e] = f32_to_bf16((x[e] - mean) * rstd * w[e] + b[e]);
}

// ---------------- generic bf16 WMMA GEMM (async double-buffered) --------------
// C[M,N] = act(A[M,K] @ W[K,N] + bias) (+ residual), W given transposed: WT[N][K].
// flags bit0 = GELU. Cf/Cb/bias/residual may be null.
#define GTM 128
#define GTN 128
#define GTK 32
__global__ __launch_bounds__(256) void k_gemm_bf16(const u16* __restrict__ A,
                                                   const u16* __restrict__ WT,
                                                   const float* __restrict__ bias,
                                                   const float* __restrict__ residual,
                                                   float* __restrict__ Cf,
                                                   u16* __restrict__ Cb,
                                                   int M, int N, int K, int flags) {
  __shared__ __align__(16) u16 As[2][GTM][GTK];   // 2 x 8 KB
  __shared__ __align__(16) u16 Bs[2][GTN][GTK];   // 2 x 8 KB
  int tid  = threadIdx.x;
  int lane = tid & 31;
  int wave = tid >> 5;        // 0..7
  int wm   = wave >> 1;       // 0..3 -> 32-row strip
  int wn   = wave & 1;        // 0..1 -> 64-col strip
  int bm   = blockIdx.y * GTM;
  int bn   = blockIdx.x * GTN;

  // Each wave issues 4 async b128 instructions per tile (2 A + 2 B).
  auto issue_tile = [&](int k0, int buf) {
#pragma unroll
    for (int t = 0; t < 2; ++t) {
      int v = tid + t * 256;
      int r = v >> 2, kc = (v & 3) << 3;
      async_g2l_b128(A  + (size_t)(bm + r) * K + k0 + kc, &As[buf][r][kc]);
      async_g2l_b128(WT + (size_t)(bn + r) * K + k0 + kc, &Bs[buf][r][kc]);
    }
  };

  v8f acc[2][4] = {};
  issue_tile(0, 0);
  for (int k0 = 0; k0 < K; k0 += GTK) {
    int buf = (k0 / GTK) & 1;
    bool nxt = (k0 + GTK) < K;
    if (nxt) {
      issue_tile(k0 + GTK, buf ^ 1);
      S_WAIT_ASYNC(0x4);                 // current tile's 4 ops drained
    } else {
      S_WAIT_ASYNC(0x0);
    }
    __syncthreads();                     // publish LDS tile to all waves
    // Load all fragments into distinct registers first so a single dscnt wait
    // covers them and the 8 WMMAs issue back-to-back.
    v16bf a0 = frag_a(&As[buf][0][0], GTK, wm * 32, 0, lane);
    v16bf a1 = frag_a(&As[buf][0][0], GTK, wm * 32 + 16, 0, lane);
    v16bf bfr[4];
#pragma unroll
    for (int tj = 0; tj < 4; ++tj)
      bfr[tj] = frag_b_n(&Bs[buf][0][0], GTK, wn * 64 + tj * 16, 0, lane);
#pragma unroll
    for (int tj = 0; tj < 4; ++tj) {
      acc[0][tj] = wmma_bf16(a0, bfr[tj], acc[0][tj]);
      acc[1][tj] = wmma_bf16(a1, bfr[tj], acc[1][tj]);
    }
    __syncthreads();                     // all reads done before buffer reuse
  }

  // Epilogue. C layout: lane holds N = lane&15, rows (lane>>4)*8 + i.
  int rofs = (lane >> 4) << 3;
  int ncol = lane & 15;
#pragma unroll
  for (int ti = 0; ti < 2; ++ti)
#pragma unroll
    for (int tj = 0; tj < 4; ++tj) {
      int nb = bn + wn * 64 + tj * 16 + ncol;
#pragma unroll
      for (int i = 0; i < 8; ++i) {
        int m = bm + wm * 32 + ti * 16 + rofs + i;
        float val = acc[ti][tj][i];
        if (bias)      val += bias[nb];
        if (flags & 1) val  = gelu_f(val);
        if (residual)  val += residual[(size_t)m * N + nb];
        if (Cf) Cf[(size_t)m * N + nb] = val;
        if (Cb) Cb[(size_t)m * N + nb] = f32_to_bf16(val);
      }
    }
}

// ---------------- flash attention: 1 wave per (b, h, 16-query tile) -----------
__global__ __launch_bounds__(32) void k_attn(const u16* __restrict__ qkv,
                                             u16* __restrict__ y, int T) {
  const int E = 1024, HD = 64, QLD = 3072;
  int qb = blockIdx.x, h = blockIdx.y, b = blockIdx.z;
  int lane = threadIdx.x;
  int q0 = qb * 16;
  size_t rbase = (size_t)b * T * QLD;
  int hofs = h * HD;

  __shared__ __align__(16) u16 Qs[16][64];
  __shared__ __align__(16) u16 Ks[2][32][64];
  __shared__ __align__(16) u16 Vs[2][32][64];
  __shared__ __align__(16) u16 Ps[16][32];

  // 16 async b128 instructions per 32-key K/V tile.
  auto issue_kv = [&](int j, int buf) {
#pragma unroll
    for (int t = 0; t < 8; ++t) {
      int v = lane + t * 32;
      int r = v >> 3, c = (v & 7) << 3;
      async_g2l_b128(qkv + rbase + (size_t)(j + r) * QLD + E     + hofs + c, &Ks[buf][r][c]);
      async_g2l_b128(qkv + rbase + (size_t)(j + r) * QLD + 2 * E + hofs + c, &Vs[buf][r][c]);
    }
  };

#pragma unroll
  for (int t = 0; t < 4; ++t) {               // Q: 16x64 = 128 x uint4
    int v = lane + t * 32;
    int r = v >> 3, c = (v & 7) << 3;
    *(uint4*)&Qs[r][c] = *(const uint4*)(qkv + rbase + (size_t)(q0 + r) * QLD + hofs + c);
  }
  issue_kv(0, 0);
  __syncthreads();
  v16bf qa0 = frag_a(&Qs[0][0], 64, 0, 0,  lane);
  v16bf qa1 = frag_a(&Qs[0][0], 64, 0, 32, lane);

  float mrow[8], lrow[8];
#pragma unroll
  for (int i = 0; i < 8; ++i) { mrow[i] = -1e30f; lrow[i] = 0.f; }
  v8f oacc[4] = {};

  int rofs = (lane >> 4) << 3;
  int ncol = lane & 15;
  const float scale = 0.125f;                  // 1/sqrt(64)

  for (int j = 0; j < q0 + 16; j += 32) {
    int buf = (j >> 5) & 1;
    bool nxt = (j + 32) < (q0 + 16);
    if (nxt) {
      issue_kv(j + 32, buf ^ 1);
      S_WAIT_ASYNC(0x10);
    } else {
      S_WAIT_ASYNC(0x0);
    }
    __syncthreads();

    v16bf kb[4];
    kb[0] = frag_b_n(&Ks[buf][0][0], 64, 0,  0,  lane);
    kb[1] = frag_b_n(&Ks[buf][0][0], 64, 0,  32, lane);
    kb[2] = frag_b_n(&Ks[buf][0][0], 64, 16, 0,  lane);
    kb[3] = frag_b_n(&Ks[buf][0][0], 64, 16, 32, lane);
    v8f s0 = {}, s1 = {};
    s0 = wmma_bf16(qa0, kb[0], s0);
    s0 = wmma_bf16(qa1, kb[1], s0);
    s1 = wmma_bf16(qa0, kb[2], s1);
    s1 = wmma_bf16(qa1, kb[3], s1);

#pragma unroll
    for (int i = 0; i < 8; ++i) {
      int qrow = q0 + rofs + i;
      float a = s0[i] * scale; if (j + ncol > qrow)      a = -1e30f;
      float c = s1[i] * scale; if (j + 16 + ncol > qrow) c = -1e30f;
      float mx = fmaxf(a, c);
#pragma unroll
      for (int off = 1; off < 16; off <<= 1) mx = fmaxf(mx, __shfl_xor(mx, off, 32));
      float mnew = fmaxf(mrow[i], mx);
      float ea = __expf(a - mnew);
      float ec = __expf(c - mnew);
      float rs = ea + ec;
#pragma unroll
      for (int off = 1; off < 16; off <<= 1) rs += __shfl_xor(rs, off, 32);
      float corr = __expf(mrow[i] - mnew);
      lrow[i] = lrow[i] * corr + rs;
      mrow[i] = mnew;
#pragma unroll
      for (int tj = 0; tj < 4; ++tj) oacc[tj][i] *= corr;
      Ps[rofs + i][ncol]      = f32_to_bf16(ea);
      Ps[rofs + i][16 + ncol] = f32_to_bf16(ec);
    }
    __syncthreads();
    v16bf pa = frag_a(&Ps[0][0], 32, 0, 0, lane);
    v16bf vb[4];
#pragma unroll
    for (int tj = 0; tj < 4; ++tj)
      vb[tj] = frag_b_k(&Vs[buf][0][0], 64, tj * 16, lane);
#pragma unroll
    for (int tj = 0; tj < 4; ++tj)
      oacc[tj] = wmma_bf16(pa, vb[tj], oacc[tj]);
    __syncthreads();
  }

#pragma unroll
  for (int tj = 0; tj < 4; ++tj)
#pragma unroll
    for (int i = 0; i < 8; ++i) {
      float o = oacc[tj][i] / lrow[i];
      size_t row = (size_t)(b * T + q0 + rofs + i);
      y[row * E + hofs + tj * 16 + ncol] = f32_to_bf16(o);
    }
}

// =============================================================================
extern "C" void kernel_launch(void* const* d_in, const int* in_sizes, int n_in,
                              void* d_out, int out_size, void* d_ws, size_t ws_size,
                              hipStream_t stream) {
  const int B = 2, T = 2048, E = 1024, L = 6, V = 32000;
  const int NT = B * T;                 // 4096 rows
  const int E3 = 3 * E, E4 = 4 * E;

  const int*   idx   = (const int*)  d_in[0];
  const float* wte   = (const float*)d_in[1];
  const float* wpe   = (const float*)d_in[2];
  const float* ln1w  = (const float*)d_in[3];
  const float* ln1b  = (const float*)d_in[4];
  const float* wqkv  = (const float*)d_in[5];
  const float* bqkv  = (const float*)d_in[6];
  const float* wpa   = (const float*)d_in[7];
  const float* bpa   = (const float*)d_in[8];
  const float* ln2w  = (const float*)d_in[9];
  const float* ln2b  = (const float*)d_in[10];
  const float* wfc   = (const float*)d_in[11];
  const float* bfc   = (const float*)d_in[12];
  const float* wpm   = (const float*)d_in[13];
  const float* bpm   = (const float*)d_in[14];
  const float* lnfw  = (const float*)d_in[15];
  const float* lnfb  = (const float*)d_in[16];
  const float* lmw   = (const float*)d_in[17];
  float* out = (float*)d_out;

  size_t cur = 0;
  auto take = [&](size_t bytes) -> void* {
    cur = (cur + 255) & ~(size_t)255;
    void* p = (char*)d_ws + cur;
    cur += bytes;
    return p;
  };
  float* x     = (float*)take((size_t)NT * E  * 4);
  u16*   hbuf  = (u16*)  take((size_t)NT * E  * 2);
  u16*   qkvb  = (u16*)  take((size_t)NT * E3 * 2);
  u16*   ybuf  = (u16*)  take((size_t)NT * E  * 2);
  u16*   fcb   = (u16*)  take((size_t)NT * E4 * 2);
  u16*   wqkvT = (u16*)  take((size_t)L * E3 * E * 2);
  u16*   wpaT  = (u16*)  take((size_t)L * E  * E * 2);
  u16*   wfcT  = (u16*)  take((size_t)L * E4 * E * 2);
  u16*   wpmT  = (u16*)  take((size_t)L * E  * E4 * 2);
  u16*   lmT   = (u16*)  take((size_t)V * E * 2);

  auto castGrid = [](int elems) { return dim3((elems + 255) / 256); };

  // ---- weight conversion (deterministic every call) ----
  for (int l = 0; l < L; ++l) {
    k_cast_tr<<<castGrid(E * E3), 256, 0, stream>>>(wqkv + (size_t)l * E * E3,
                                                    wqkvT + (size_t)l * E3 * E, E, E3);
    k_cast_tr<<<castGrid(E * E),  256, 0, stream>>>(wpa  + (size_t)l * E * E,
                                                    wpaT  + (size_t)l * E * E,  E, E);
    k_cast_tr<<<castGrid(E * E4), 256, 0, stream>>>(wfc  + (size_t)l * E * E4,
                                                    wfcT  + (size_t)l * E4 * E, E, E4);
    k_cast_tr<<<castGrid(E4 * E), 256, 0, stream>>>(wpm  + (size_t)l * E4 * E,
                                                    wpmT  + (size_t)l * E * E4, E4, E);
  }
  k_cast_tr<<<castGrid(E * V), 256, 0, stream>>>(lmw, lmT, E, V);

  // ---- embedding ----
  k_embed<<<castGrid(NT * E), 256, 0, stream>>>(idx, wte, wpe, x, T, E, NT * E);

  // ---- transformer layers ----
  for (int l = 0; l < L; ++l) {
    k_layernorm<<<NT, 256, 0, stream>>>(x, ln1w + (size_t)l * E, ln1b + (size_t)l * E, hbuf, E);

    // qkv = ln1(x) @ wqkv + bqkv  -> bf16
    k_gemm_bf16<<<dim3(E3 / GTN, NT / GTM), 256, 0, stream>>>(
        hbuf, wqkvT + (size_t)l * E3 * E, bqkv + (size_t)l * E3,
        nullptr, nullptr, qkvb, NT, E3, E, 0);

    // attention -> y (bf16)
    k_attn<<<dim3(T / 16, 16, B), 32, 0, stream>>>(qkvb, ybuf, T);

    // x = x + y @ wproj_attn + bproj_attn
    k_gemm_bf16<<<dim3(E / GTN, NT / GTM), 256, 0, stream>>>(
        ybuf, wpaT + (size_t)l * E * E, bpa + (size_t)l * E,
        x, x, nullptr, NT, E, E, 0);

    k_layernorm<<<NT, 256, 0, stream>>>(x, ln2w + (size_t)l * E, ln2b + (size_t)l * E, hbuf, E);

    // fc = gelu(ln2(x) @ wfc + bfc) -> bf16
    k_gemm_bf16<<<dim3(E4 / GTN, NT / GTM), 256, 0, stream>>>(
        hbuf, wfcT + (size_t)l * E4 * E, bfc + (size_t)l * E4,
        nullptr, nullptr, fcb, NT, E4, E, 1);

    // x = x + fc @ wproj_mlp + bproj_mlp
    k_gemm_bf16<<<dim3(E / GTN, NT / GTM), 256, 0, stream>>>(
        fcb, wpmT + (size_t)l * E * E4, bpm + (size_t)l * E,
        x, x, nullptr, NT, E, E4, 0);
  }

  // ---- final LN + LM head ----
  k_layernorm<<<NT, 256, 0, stream>>>(x, lnfw, lnfb, hbuf, E);
  k_gemm_bf16<<<dim3(V / GTN, NT / GTM), 256, 0, stream>>>(
      hbuf, lmT, nullptr, nullptr, out, nullptr, NT, V, E, 0);
}
